// MoELayer_32753420599442
// MI455X (gfx1250) — matrix operand order repaired and checked by reading, not compiled, not run on hardware
//
#include <hip/hip_runtime.h>

// ---- problem constants (B=8,S=2048,D=256,F=1024,E=8,K=2) ----
#define NTOK 16384
#define DDIM 256
#define FDIM 1024
#define NEXP 8
#define FC   64     // F-chunk per staging round
#define MT   32     // token tile (2 x WMMA M)

typedef __attribute__((ext_vector_type(16))) __bf16 v16bf;
typedef __attribute__((ext_vector_type(8)))  float  v8f;
typedef __attribute__((ext_vector_type(4)))  unsigned uint4v;
typedef __attribute__((ext_vector_type(8)))  int    int8v;
typedef __attribute__((ext_vector_type(4)))  int    int4v;

// dynamic-LDS partition (no static __shared__ in the FFN kernel => offsets from 0)
#define XT_OFF  0                       // 32*256 bf16 = 16384 B  row-major [m][k]
#define WT_OFF  16384                   // 64*256 bf16 = 32768 B  transposed [n][k]
#define HS_OFF  49152                   // 32*64  bf16 =  4096 B  row-major [m][n]
#define TOK_OFF 53248                   // 32 * int
#define WSC_OFF 53376                   // 32 * float
#define KR_OFF  53504                   // 32 * int
#define VS_OFF  53632                   // 32 * int
#define SMEM_BYTES 53760

__device__ __forceinline__ unsigned short f2bf(float f) {
  unsigned u = __float_as_uint(f);
  u += 0x7FFFu + ((u >> 16) & 1u);          // round-to-nearest-even
  return (unsigned short)(u >> 16);
}

__device__ __forceinline__ float gelu_exact(float x) {
  return 0.5f * x * (1.0f + erff(x * 0.70710678118654752f));
}

// ---------------- f32 -> bf16 conversion ----------------
__global__ void cvt_bf16_kernel(const float* __restrict__ src,
                                unsigned short* __restrict__ dst, int n) {
  int i = blockIdx.x * blockDim.x + threadIdx.x;
  if (i < n) dst[i] = f2bf(src[i]);
}

__global__ void zero_cnt_kernel(int* cnt) {
  if (threadIdx.x < NEXP) cnt[threadIdx.x] = 0;
}

// ---------------- router: softmax, top-2, renorm, compaction ----------------
__global__ void router_kernel(const float* __restrict__ x, const float* __restrict__ Wr,
                              float* __restrict__ gate, int* __restrict__ lists,
                              int* __restrict__ cnt) {
  __shared__ float wr[DDIM * NEXP];
  const int tid = threadIdx.x;
  for (int i = tid; i < DDIM * NEXP; i += 256) wr[i] = Wr[i];
  __syncthreads();

  const int t = blockIdx.x * 256 + tid;
  const float* xr = x + (size_t)t * DDIM;
  float l[NEXP];
#pragma unroll
  for (int e = 0; e < NEXP; ++e) l[e] = 0.f;
  for (int d = 0; d < DDIM; ++d) {
    float xv = xr[d];
#pragma unroll
    for (int e = 0; e < NEXP; ++e) l[e] = fmaf(xv, wr[d * NEXP + e], l[e]);
  }
  float m = l[0];
#pragma unroll
  for (int e = 1; e < NEXP; ++e) m = fmaxf(m, l[e]);
  float p[NEXP];
#pragma unroll
  for (int e = 0; e < NEXP; ++e) p[e] = __expf(l[e] - m);

  int i0 = 0;
#pragma unroll
  for (int e = 1; e < NEXP; ++e) if (p[e] > p[i0]) i0 = e;   // first occurrence (matches top_k)
  int i1 = (i0 == 0) ? 1 : 0;
#pragma unroll
  for (int e = 0; e < NEXP; ++e) if (e != i0 && p[e] > p[i1]) i1 = e;

  const float s2 = p[i0] + p[i1];
  const float w0 = p[i0] / s2, w1 = p[i1] / s2;
#pragma unroll
  for (int e = 0; e < NEXP; ++e)
    gate[(size_t)t * NEXP + e] = (e == i0) ? w0 : ((e == i1) ? w1 : 0.f);

  int pos0 = atomicAdd(&cnt[i0], 1);
  lists[i0 * NTOK + pos0] = t;               // rank 0
  int pos1 = atomicAdd(&cnt[i1], 1);
  lists[i1 * NTOK + pos1] = t | (1 << 20);   // rank 1
}

// ---- TDM gather-mode load of 16 token rows (bf16, 256 wide) into LDS ----
__device__ __forceinline__ void tdm_gather16(const unsigned short* xb, unsigned lds_addr,
                                             const int* tokS, int base) {
  unsigned long long ga = (unsigned long long)xb;   // byte address of tensor start
  uint4v g0;
  g0[0] = 0x80000001u;                              // count=1 | gather_mode | 16-bit idx
  g0[1] = lds_addr;                                 // LDS byte address of tile
  g0[2] = (unsigned)(ga & 0xFFFFFFFFu);             // global_addr[31:0]
  g0[3] = (unsigned)((ga >> 32) & 0x01FFFFFFu) | 0x80000000u;  // ga[56:32] | type=2
  int8v g1;
  g1[0] = (int)(1u << 16);                          // wg_mask=0, data_size=1 (2 bytes)
  g1[1] = (int)((unsigned)DDIM << 16);              // tensor_dim0[15:0]
  g1[2] = (int)((unsigned)NTOK << 16);              // td0 hi=0 | tensor_dim1[15:0]
  g1[3] = (int)((unsigned)DDIM << 16);              // td1 hi=0 | tile_dim0
  g1[4] = 16;                                       // tile_dim1 = #indices, tile_dim2=0
  g1[5] = DDIM;                                     // tensor_dim0_stride[31:0]
  g1[6] = 0;                                        // stride0 hi | stride1 lo (ignored)
  g1[7] = 0;
  int4v g2, g3;
#pragma unroll
  for (int j = 0; j < 4; ++j)
    g2[j] = (tokS[base + 2 * j] & 0xFFFF) | (tokS[base + 2 * j + 1] << 16);
#pragma unroll
  for (int j = 0; j < 4; ++j)
    g3[j] = (tokS[base + 8 + 2 * j] & 0xFFFF) | (tokS[base + 8 + 2 * j + 1] << 16);
#if defined(__clang_major__) && __clang_major__ >= 23
  int8v gz = (int8v){0, 0, 0, 0, 0, 0, 0, 0};
  __builtin_amdgcn_tensor_load_to_lds(g0, g1, g2, g3, gz, 0);
#else
  __builtin_amdgcn_tensor_load_to_lds(g0, g1, g2, g3, 0);
#endif
}

// ---------------- fused 2-layer FFN with WMMA (shared + gathered experts) ----------------
__global__ __launch_bounds__(256)
void moe_ffn_kernel(const unsigned short* __restrict__ xb,
                    const unsigned short* __restrict__ w1s, const unsigned short* __restrict__ w2s,
                    const float* __restrict__ b1s, const float* __restrict__ b2s,
                    const unsigned short* __restrict__ w1p, const unsigned short* __restrict__ w2p,
                    const float* __restrict__ b1p, const float* __restrict__ b2p,
                    const float* __restrict__ gate, const int* __restrict__ lists,
                    const int* __restrict__ cnt,
                    float* __restrict__ out, float* __restrict__ c0, float* __restrict__ c1) {
  extern __shared__ uint4v smem_v[];        // 16-byte aligned dynamic LDS
  char* smem = (char*)smem_v;
  unsigned short* Xt = (unsigned short*)(smem + XT_OFF);
  unsigned short* Wt = (unsigned short*)(smem + WT_OFF);
  unsigned short* Hs = (unsigned short*)(smem + HS_OFF);
  int*   tokS = (int*)(smem + TOK_OFF);
  float* wS   = (float*)(smem + WSC_OFF);
  int*   krS  = (int*)(smem + KR_OFF);
  int*   vS   = (int*)(smem + VS_OFF);

  const int slot = blockIdx.y;       // 0 = shared expert, 1..8 = private experts
  const int tile = blockIdx.x;
  const int expert = slot - 1;

  const int count = (slot == 0) ? NTOK : cnt[expert];
  if (tile * MT >= count) return;    // whole block exits: no partial-EXEC WMMA

  const int tid  = threadIdx.x;
  const int lane = tid & 31;
  const int wave = tid >> 5;        // 0..7
  const int mt   = wave >> 2;       // m-tile: 0/1
  const int nq   = wave & 3;        // n-quarter: 0..3
  const int lh   = lane >> 4;       // lane half
  const int ln   = lane & 15;

  // ---- gather tile metadata
  if (tid < MT) {
    int idx = tile * MT + tid;
    int tok = 0, kr = 0, valid = 0; float w = 0.f;
    if (slot == 0) { tok = idx; w = 1.f; valid = 1; }
    else if (idx < count) {
      int ent = lists[expert * NTOK + idx];
      tok = ent & 0xFFFFF; kr = ent >> 20;
      w = gate[(size_t)tok * NEXP + expert];
      valid = 1;
    }
    tokS[tid] = tok; wS[tid] = w; krS[tid] = kr; vS[tid] = valid;
  }
  __syncthreads();

  // ---- TDM gather: 32 token rows of bf16 x into LDS (two 16-index descriptors)
  if (wave == 0) {
    tdm_gather16(xb, XT_OFF, tokS, 0);
    tdm_gather16(xb, XT_OFF + 16 * DDIM * 2, tokS, 16);
    __builtin_amdgcn_s_wait_tensorcnt(0);
  }
  // visibility to other waves via the loop-top barrier below

  const unsigned short* W1 = (slot == 0) ? w1s : (w1p + (size_t)expert * DDIM * FDIM);
  const unsigned short* W2 = (slot == 0) ? w2s : (w2p + (size_t)expert * FDIM * DDIM);
  const float* b1 = (slot == 0) ? b1s : (b1p + expert * FDIM);
  const float* b2 = (slot == 0) ? b2s : (b2p + expert * DDIM);

  v8f accO[4];
#pragma unroll
  for (int i = 0; i < 4; ++i) accO[i] = (v8f){0, 0, 0, 0, 0, 0, 0, 0};

  union Frag { v16bf v; uint4v q[2]; };

  for (int fc = 0; fc < FDIM; fc += FC) {
    __syncthreads();
    // stage W1 chunk transposed: element (k, fc+n) -> Wt[n*DDIM + k]
    for (int i = tid; i < (DDIM * FC) / 2; i += 256) {
      int k = i >> 5;                // 32 dwords per k-row of the 64-wide chunk
      int n = (i & 31) * 2;
      unsigned v = *(const unsigned*)(W1 + (size_t)k * FDIM + fc + n);
      Wt[n * DDIM + k]       = (unsigned short)(v & 0xFFFFu);
      Wt[(n + 1) * DDIM + k] = (unsigned short)(v >> 16);
    }
    __syncthreads();

    // GEMM1: this wave computes H tile (mt, n-range [16*nq, 16*nq+16))
    v8f accH = (v8f){0, 0, 0, 0, 0, 0, 0, 0};
    const uint4v* arow = (const uint4v*)(Xt + (16 * mt + ln) * DDIM);  // 128B-aligned row
    const uint4v* bcol = (const uint4v*)(Wt + (16 * nq + ln) * DDIM);
#pragma unroll
    for (int k0 = 0; k0 < DDIM; k0 += 32) {
      Frag A, Bm;
      int aq = (k0 >> 3) + lh;           // A: dword groups {ab..ab+3},{ab+8..ab+11}, ab=4*aq
      A.q[0] = arow[aq];
      A.q[1] = arow[aq + 2];
      int bq = (k0 >> 3) + lh;           // B: same dword groups as A, bb=4*bq
      Bm.q[0] = bcol[bq];
      Bm.q[1] = bcol[bq + 2];
      accH = __builtin_amdgcn_wmma_f32_16x16x32_bf16(false, A.v, false, Bm.v,
                                                     (short)0, accH, false, false);
    }
    // epilogue: bias + exact GELU -> Hs bf16  (C/D layout: M = r + 8*lh, N = ln)
    float b1v = b1[fc + 16 * nq + ln];
#pragma unroll
    for (int r = 0; r < 8; ++r) {
      int m = 16 * mt + 8 * lh + r;
      Hs[m * FC + 16 * nq + ln] = f2bf(gelu_exact(accH[r] + b1v));
    }
    __syncthreads();

    // restage same buffer with W2 chunk transposed: (fc+k, n) -> Wt[n*FC + k]
    for (int i = tid; i < (FC * DDIM) / 2; i += 256) {
      int k = i >> 7;                // 128 dwords per k-row (256 cols)
      int n = (i & 127) * 2;
      unsigned v = *(const unsigned*)(W2 + (size_t)(fc + k) * DDIM + n);
      Wt[n * FC + k]       = (unsigned short)(v & 0xFFFFu);
      Wt[(n + 1) * FC + k] = (unsigned short)(v >> 16);
    }
    __syncthreads();

    // GEMM2: accO[nt] += H(16x64) @ W2chunk(64x16), n = 64*nq + 16*nt + ln
    const uint4v* hrow = (const uint4v*)(Hs + (16 * mt + ln) * FC);    // 128B-aligned row
#pragma unroll
    for (int nt = 0; nt < 4; ++nt) {
      const uint4v* b2col = (const uint4v*)(Wt + (64 * nq + 16 * nt + ln) * FC);
#pragma unroll
      for (int k0 = 0; k0 < FC; k0 += 32) {
        Frag A, Bm;
        int aq = (k0 >> 3) + lh;
        A.q[0] = hrow[aq];
        A.q[1] = hrow[aq + 2];
        int bq = (k0 >> 3) + lh;
        Bm.q[0] = b2col[bq];
        Bm.q[1] = b2col[bq + 2];
        accO[nt] = __builtin_amdgcn_wmma_f32_16x16x32_bf16(false, A.v, false, Bm.v,
                                                           (short)0, accO[nt], false, false);
      }
    }
  }

  // ---- epilogue: +b2, gate scale, scatter (each (token,rank) written exactly once)
#pragma unroll
  for (int nt = 0; nt < 4; ++nt) {
    int n = 64 * nq + 16 * nt + ln;
    float b2v = b2[n];
#pragma unroll
    for (int r = 0; r < 8; ++r) {
      int m = 16 * mt + 8 * lh + r;
      if (!vS[m]) continue;
      float val = accO[nt][r] + b2v;
      size_t o = (size_t)tokS[m] * DDIM + n;
      if (slot == 0) out[o] = val;
      else (krS[m] ? c1 : c0)[o] = val * wS[m];
    }
  }
}

// ---------------- final deterministic combine ----------------
__global__ void combine_kernel(float* __restrict__ out,
                               const float* __restrict__ c0, const float* __restrict__ c1) {
  int i = blockIdx.x * 256 + threadIdx.x;
  out[i] += c0[i] + c1[i];
}

extern "C" void kernel_launch(void* const* d_in, const int* in_sizes, int n_in,
                              void* d_out, int out_size, void* d_ws, size_t ws_size,
                              hipStream_t stream) {
  const float* x   = (const float*)d_in[0];
  const float* Wr  = (const float*)d_in[1];
  const float* W1s = (const float*)d_in[2];
  const float* b1s = (const float*)d_in[3];
  const float* W2s = (const float*)d_in[4];
  const float* b2s = (const float*)d_in[5];
  const float* W1p = (const float*)d_in[6];
  const float* b1p = (const float*)d_in[7];
  const float* W2p = (const float*)d_in[8];
  const float* b2p = (const float*)d_in[9];
  float* out = (float*)d_out;

  char* ws = (char*)d_ws;
  size_t off = 0;
  auto wsalloc = [&](size_t bytes) {
    void* p = ws + off;
    off += (bytes + 255) & ~(size_t)255;
    return p;
  };
  unsigned short* xb   = (unsigned short*)wsalloc((size_t)NTOK * DDIM * 2);
  unsigned short* w1sb = (unsigned short*)wsalloc((size_t)DDIM * FDIM * 2);
  unsigned short* w2sb = (unsigned short*)wsalloc((size_t)FDIM * DDIM * 2);
  unsigned short* w1pb = (unsigned short*)wsalloc((size_t)NEXP * DDIM * FDIM * 2);
  unsigned short* w2pb = (unsigned short*)wsalloc((size_t)NEXP * FDIM * DDIM * 2);
  float* gate = (float*)wsalloc((size_t)NTOK * NEXP * 4);
  int*   cnt  = (int*)wsalloc(NEXP * 4);
  int*   lists= (int*)wsalloc((size_t)NEXP * NTOK * 4);
  float* c0   = (float*)wsalloc((size_t)NTOK * DDIM * 4);
  float* c1   = (float*)wsalloc((size_t)NTOK * DDIM * 4);

  // 1) precision conversion (f32 -> bf16)
  {
    int n;
    n = NTOK * DDIM;
    cvt_bf16_kernel<<<(n + 255) / 256, 256, 0, stream>>>(x, xb, n);
    n = DDIM * FDIM;
    cvt_bf16_kernel<<<(n + 255) / 256, 256, 0, stream>>>(W1s, w1sb, n);
    n = FDIM * DDIM;
    cvt_bf16_kernel<<<(n + 255) / 256, 256, 0, stream>>>(W2s, w2sb, n);
    n = NEXP * DDIM * FDIM;
    cvt_bf16_kernel<<<(n + 255) / 256, 256, 0, stream>>>(W1p, w1pb, n);
    n = NEXP * FDIM * DDIM;
    cvt_bf16_kernel<<<(n + 255) / 256, 256, 0, stream>>>(W2p, w2pb, n);
  }

  // 2) router + compaction
  zero_cnt_kernel<<<1, 32, 0, stream>>>(cnt);
  router_kernel<<<NTOK / 256, 256, 0, stream>>>(x, Wr, gate, lists, cnt);

  // 3) fused FFN: slot 0 = shared (all tokens), slots 1..8 = gathered experts
  dim3 grid(NTOK / MT, NEXP + 1);
  moe_ffn_kernel<<<grid, 256, SMEM_BYTES, stream>>>(xb, w1sb, w2sb, b1s, b2s,
                                                    w1pb, w2pb, b1p, b2p,
                                                    gate, lists, cnt, out, c0, c1);

  // 4) deterministic combine
  combine_kernel<<<(NTOK * DDIM) / 256, 256, 0, stream>>>(out, c0, c1);
}
